// GraphConvResidualNet_21345987461151
// MI455X (gfx1250) — matrix operand order repaired and checked by reading, not compiled, not run on hardware
//
#include <hip/hip_runtime.h>
#include <hip/hip_bf16.h>

typedef __attribute__((ext_vector_type(16))) __bf16 v16bf;
typedef __attribute__((ext_vector_type(8)))  float  v8f;

#define FD 128   // F == D == 128
#define NCLS 10
#define NLAYERS 4

// ---------------- utility: zero float4 buffer ----------------
__global__ void zero4_kernel(float4* __restrict__ p, int n4) {
    int i = blockIdx.x * blockDim.x + threadIdx.x;
    if (i < n4) p[i] = make_float4(0.f, 0.f, 0.f, 0.f);
}

// ---------------- pack 128x128 fp32 weight into WMMA-B bf16 fragment layout ----------------
// Fragment (k0,nt): lane L<16 holds N=nt*16+L, K=k0+{0..7,16..23} ; lane L>=16 holds N=nt*16+(L&15), K=k0+{8..15,24..31}.
// Packed index: ((k0/32)*8 + nt)*512 + lane*16 + j   (16 contiguous bf16 = 32B per lane)
__global__ void pack_w_bf16(const float* __restrict__ W, __bf16* __restrict__ P) {
    int id = blockIdx.x * blockDim.x + threadIdx.x;   // 16384 total
    int frag = id >> 9;
    int lane = (id >> 4) & 31;
    int j    = id & 15;
    int k0 = (frag >> 3) * 32;
    int nt = frag & 7;
    int k = k0 + ((lane >> 4) ? 8 : 0) + (j & 7) + ((j >> 3) * 16);
    int n = nt * 16 + (lane & 15);
    P[id] = (__bf16)W[k * FD + n];
}

// ---------------- edge scatter: agg[dst] += h[src] * w ; one wave32 per edge ----------------
__global__ void scatter_kernel(const long long* __restrict__ ei, const float* __restrict__ ew,
                               const float* __restrict__ H, float* __restrict__ Agg, int E) {
    int e    = (blockIdx.x * blockDim.x + threadIdx.x) >> 5;
    int lane = threadIdx.x & 31;
    if (e >= E) return;
    long long s = ei[e];
    long long d = ei[(size_t)E + e];
    float w = ew[e];
    float4 v = ((const float4*)(H + (size_t)s * FD))[lane];
    float* ap = Agg + (size_t)d * FD + lane * 4;
    atomicAdd(ap + 0, v.x * w);
    atomicAdd(ap + 1, v.y * w);
    atomicAdd(ap + 2, v.z * w);
    atomicAdd(ap + 3, v.w * w);
}

// ---------------- dual GEMM: Hout = Agg@Wrel + Hin@Wroot + bias ; accumulate BN col stats ----------------
// Block = 8 waves, each wave -> 16 rows x 128 cols (8 accumulators of 16x16 f32).
__global__ void __launch_bounds__(256)
gemm_dual_kernel(const float* __restrict__ Agg, const float* __restrict__ Hin,
                 const __bf16* __restrict__ Prel, const __bf16* __restrict__ Proot,
                 const float* __restrict__ bias, float* __restrict__ Hout,
                 float* __restrict__ colsum, float* __restrict__ colsq, int nrows) {
    __shared__ float s_sum[FD];
    __shared__ float s_sq[FD];
    int tid = threadIdx.x;
    if (tid < FD) { s_sum[tid] = 0.f; s_sq[tid] = 0.f; }
    __syncthreads();

    int lane = tid & 31;
    int wave = tid >> 5;
    int rowBase = blockIdx.x * 128 + wave * 16;
    int r = rowBase + (lane & 15);
    bool rowOK = (r < nrows);
    int khi = (lane >> 4) * 8;   // lanes 16-31 hold K halves offset by 8 (and +24)

    v8f acc[8] = {};

    for (int k0 = 0; k0 < FD; k0 += 32) {
        v16bf aAgg = {};
        v16bf aH   = {};
        if (rowOK) {
            const float* pa = Agg + (size_t)r * FD + k0 + khi;
            const float* ph = Hin + (size_t)r * FD + k0 + khi;
            #pragma unroll
            for (int j = 0; j < 8; ++j) {
                aAgg[j]     = (__bf16)pa[j];
                aAgg[j + 8] = (__bf16)pa[j + 16];
                aH[j]       = (__bf16)ph[j];
                aH[j + 8]   = (__bf16)ph[j + 16];
            }
        }
        int fragBase = (k0 >> 5) * 8;
        #pragma unroll
        for (int nt = 0; nt < 8; ++nt) {
            v16bf bRel  = *(const v16bf*)(Prel  + ((size_t)(fragBase + nt) * 32 + lane) * 16);
            v16bf bRoot = *(const v16bf*)(Proot + ((size_t)(fragBase + nt) * 32 + lane) * 16);
            acc[nt] = __builtin_amdgcn_wmma_f32_16x16x32_bf16(false, aAgg, false, bRel,
                                                              (short)0, acc[nt], false, false);
            acc[nt] = __builtin_amdgcn_wmma_f32_16x16x32_bf16(false, aH, false, bRoot,
                                                              (short)0, acc[nt], false, false);
        }
    }

    // bias add, store, BN partial stats. C layout: element v of acc: M = rowBase + v + 8*(lane>=16), N = nt*16 + lane%16
    int colLo = lane & 15;
    int rowHalf = (lane >> 4) * 8;
    #pragma unroll
    for (int nt = 0; nt < 8; ++nt) {
        int col = nt * 16 + colLo;
        float b = bias[col];
        float psum = 0.f, psq = 0.f;
        #pragma unroll
        for (int v = 0; v < 8; ++v) {
            int row = rowBase + v + rowHalf;
            float val = acc[nt][v] + b;
            if (row < nrows) {
                Hout[(size_t)row * FD + col] = val;
                psum += val;
                psq  += val * val;
            }
        }
        atomicAdd(&s_sum[col], psum);
        atomicAdd(&s_sq[col],  psq);
    }
    __syncthreads();
    if (tid < FD) {
        atomicAdd(&colsum[tid], s_sum[tid]);
        atomicAdd(&colsq[tid],  s_sq[tid]);
    }
}

// ---------------- BN stats -> per-channel scale/shift ----------------
__global__ void bnstats_kernel(const float* __restrict__ colsum, const float* __restrict__ colsq,
                               const float* __restrict__ gamma, const float* __restrict__ beta,
                               float* __restrict__ scale, float* __restrict__ shift, int n) {
    int c = threadIdx.x;
    float inv_n = 1.0f / (float)n;
    float mean = colsum[c] * inv_n;
    float var  = colsq[c] * inv_n - mean * mean;
    float sc   = gamma[c] * rsqrtf(var + 1e-5f);
    scale[c] = sc;
    shift[c] = beta[c] - mean * sc;
}

// ---------------- apply BN + ReLU in place ----------------
__global__ void bnapply_kernel(float* __restrict__ H, const float* __restrict__ scale,
                               const float* __restrict__ shift, int n4) {
    int i = blockIdx.x * blockDim.x + threadIdx.x;
    if (i >= n4) return;
    float4 v = ((float4*)H)[i];
    int cb = (i & 31) * 4;   // 128 cols = 32 float4 per row
    v.x = fmaxf(0.f, v.x * scale[cb + 0] + shift[cb + 0]);
    v.y = fmaxf(0.f, v.y * scale[cb + 1] + shift[cb + 1]);
    v.z = fmaxf(0.f, v.z * scale[cb + 2] + shift[cb + 2]);
    v.w = fmaxf(0.f, v.w * scale[cb + 3] + shift[cb + 3]);
    ((float4*)H)[i] = v;
}

// ---------------- graph pooling: g[batch[n]] += h[n] ; one wave per node ----------------
__global__ void pool_kernel(const long long* __restrict__ batch, const float* __restrict__ H,
                            float* __restrict__ Gacc, int N) {
    int nidx = (blockIdx.x * blockDim.x + threadIdx.x) >> 5;
    int lane = threadIdx.x & 31;
    if (nidx >= N) return;
    long long b = batch[nidx];
    float4 v = ((const float4*)(H + (size_t)nidx * FD))[lane];
    float* gp = Gacc + (size_t)b * FD + lane * 4;
    atomicAdd(gp + 0, v.x);
    atomicAdd(gp + 1, v.y);
    atomicAdd(gp + 2, v.z);
    atomicAdd(gp + 3, v.w);
}

// ---------------- MLP1: G2 = relu(G @ lin1_w + b1) via WMMA ----------------
__global__ void __launch_bounds__(256)
mlp1_kernel(const float* __restrict__ Gin, const __bf16* __restrict__ P,
            const float* __restrict__ b1, float* __restrict__ G2, int rows) {
    int tid = threadIdx.x;
    int lane = tid & 31;
    int wave = tid >> 5;
    int rowBase = blockIdx.x * 128 + wave * 16;
    int r = rowBase + (lane & 15);
    bool rowOK = (r < rows);
    int khi = (lane >> 4) * 8;

    v8f acc[8] = {};
    for (int k0 = 0; k0 < FD; k0 += 32) {
        v16bf a = {};
        if (rowOK) {
            const float* pg = Gin + (size_t)r * FD + k0 + khi;
            #pragma unroll
            for (int j = 0; j < 8; ++j) {
                a[j]     = (__bf16)pg[j];
                a[j + 8] = (__bf16)pg[j + 16];
            }
        }
        int fragBase = (k0 >> 5) * 8;
        #pragma unroll
        for (int nt = 0; nt < 8; ++nt) {
            v16bf bb = *(const v16bf*)(P + ((size_t)(fragBase + nt) * 32 + lane) * 16);
            acc[nt] = __builtin_amdgcn_wmma_f32_16x16x32_bf16(false, a, false, bb,
                                                              (short)0, acc[nt], false, false);
        }
    }
    int colLo = lane & 15;
    int rowHalf = (lane >> 4) * 8;
    #pragma unroll
    for (int nt = 0; nt < 8; ++nt) {
        int col = nt * 16 + colLo;
        float b = b1[col];
        #pragma unroll
        for (int v = 0; v < 8; ++v) {
            int row = rowBase + v + rowHalf;
            if (row < rows)
                G2[(size_t)row * FD + col] = fmaxf(0.f, acc[nt][v] + b);
        }
    }
}

// ---------------- head: logits = G2 @ lin2_w + b2 ; log_softmax ----------------
__global__ void head_kernel(const float* __restrict__ G2, const float* __restrict__ W2,
                            const float* __restrict__ b2, float* __restrict__ out, int G) {
    int row = blockIdx.x * blockDim.x + threadIdx.x;
    if (row >= G) return;
    float logits[NCLS];
    #pragma unroll
    for (int c = 0; c < NCLS; ++c) logits[c] = b2[c];
    const float* g = G2 + (size_t)row * FD;
    for (int k = 0; k < FD; ++k) {
        float gv = g[k];
        const float* wr = W2 + (size_t)k * NCLS;
        #pragma unroll
        for (int c = 0; c < NCLS; ++c) logits[c] += gv * wr[c];
    }
    float m = logits[0];
    #pragma unroll
    for (int c = 1; c < NCLS; ++c) m = fmaxf(m, logits[c]);
    float s = 0.f;
    #pragma unroll
    for (int c = 0; c < NCLS; ++c) s += expf(logits[c] - m);
    float lse = m + logf(s);
    #pragma unroll
    for (int c = 0; c < NCLS; ++c) out[(size_t)row * NCLS + c] = logits[c] - lse;
}

extern "C" void kernel_launch(void* const* d_in, const int* in_sizes, int n_in,
                              void* d_out, int out_size, void* d_ws, size_t ws_size,
                              hipStream_t stream) {
    const float*     x      = (const float*)d_in[0];
    const long long* ei     = (const long long*)d_in[1];
    const long long* batch  = (const long long*)d_in[2];
    const float*     ew     = (const float*)d_in[3];
    const float*     W_root = (const float*)d_in[4];
    const float*     W_rel  = (const float*)d_in[5];
    const float*     b_conv = (const float*)d_in[6];
    const float*     gamma  = (const float*)d_in[7];
    const float*     beta   = (const float*)d_in[8];
    const float*     lin1_w = (const float*)d_in[9];
    const float*     lin1_b = (const float*)d_in[10];
    const float*     lin2_w = (const float*)d_in[11];
    const float*     lin2_b = (const float*)d_in[12];
    float* out = (float*)d_out;

    const int N = in_sizes[0] / FD;
    const int E = in_sizes[3];
    const int G = out_size / NCLS;

    // workspace carve-out
    char* ws = (char*)d_ws;
    size_t off = 0;
    auto carve = [&](size_t bytes) -> void* {
        void* p = ws + off;
        off = (off + bytes + 255) & ~(size_t)255;
        return p;
    };
    float*  hbuf   = (float*)carve((size_t)N * FD * sizeof(float));
    float*  agg    = (float*)carve((size_t)N * FD * sizeof(float));
    float*  gpool  = (float*)carve((size_t)G * FD * sizeof(float));
    float*  g2     = (float*)carve((size_t)G * FD * sizeof(float));
    __bf16* pRel   = (__bf16*)carve(FD * FD * sizeof(__bf16));
    __bf16* pRoot  = (__bf16*)carve(FD * FD * sizeof(__bf16));
    __bf16* pLin   = (__bf16*)carve(FD * FD * sizeof(__bf16));
    float*  colsum = (float*)carve(2 * FD * sizeof(float)); // colsum + colsq contiguous
    float*  colsq  = colsum + FD;
    float*  bnscale = (float*)carve(FD * sizeof(float));
    float*  bnshift = (float*)carve(FD * sizeof(float));
    (void)n_in; (void)ws_size;

    const int nGemmBlocks = (N + 127) / 128;
    const float* hin = x;

    for (int l = 0; l < NLAYERS; ++l) {
        zero4_kernel<<<(N * 32 + 255) / 256, 256, 0, stream>>>((float4*)agg, N * 32);
        zero4_kernel<<<1, 256, 0, stream>>>((float4*)colsum, (2 * FD) / 4);
        pack_w_bf16<<<64, 256, 0, stream>>>(W_rel + (size_t)l * FD * FD, pRel);
        pack_w_bf16<<<64, 256, 0, stream>>>(W_root + (size_t)l * FD * FD, pRoot);
        scatter_kernel<<<(E * 32 + 255) / 256, 256, 0, stream>>>(ei, ew, hin, agg, E);
        gemm_dual_kernel<<<nGemmBlocks, 256, 0, stream>>>(agg, hin, pRel, pRoot,
                                                          b_conv + l * FD, hbuf,
                                                          colsum, colsq, N);
        bnstats_kernel<<<1, FD, 0, stream>>>(colsum, colsq, gamma + l * FD, beta + l * FD,
                                             bnscale, bnshift, N);
        bnapply_kernel<<<(N * 32 + 255) / 256, 256, 0, stream>>>(hbuf, bnscale, bnshift, N * 32);
        hin = hbuf;
    }

    zero4_kernel<<<(G * 32 + 255) / 256, 256, 0, stream>>>((float4*)gpool, G * 32);
    pool_kernel<<<(N * 32 + 255) / 256, 256, 0, stream>>>(batch, hbuf, gpool, N);
    pack_w_bf16<<<64, 256, 0, stream>>>(lin1_w, pLin);
    mlp1_kernel<<<(G + 127) / 128, 256, 0, stream>>>(gpool, pLin, lin1_b, g2, G);
    head_kernel<<<(G + 255) / 256, 256, 0, stream>>>(g2, lin2_w, lin2_b, out, G);
}
